// MultiHeadAttention_65292092834426
// MI455X (gfx1250) — matrix operand order repaired and checked by reading, not compiled, not run on hardware
//
#include <hip/hip_runtime.h>

// ---------------------------------------------------------------------------
// MultiHeadAttention forward for MI455X (gfx1250, wave32, WMMA + TDM).
// B=4, S=2048, D=1024, H=16, HD=64.  fp32 in/out; bf16 WMMA internally
// (v_wmma_f32_16x16x32_bf16, f32 accumulation).  Flash-style fused attention.
// Pipeline:
//   0) one-shot fp32->bf16 conversion of x and all weights (native cvt)
//   1) QKV GEMM: A = bf16 x (b128 loads), B = weight tile staged into LDS by
//      the Tensor Data Mover (tensor_load_to_lds, double-buffered,
//      s_wait_tensorcnt), bias+scale fused, head-major bf16 outputs (V^T)
//   2) fused flash attention (online softmax, P reshaped C->A via LDS)
//   3) output projection (same TDM-staged weight path), fp32 out
// Workspace: xb 16.8MB + w*4 8.4MB + q/k/vT 50.3MB + ctx 16.8MB = 92.3 MB.
// ---------------------------------------------------------------------------

#define Bc  4
#define Sc  2048
#define Dc  1024
#define Hc  16
#define HDc 64
#define SCALEF 0.125f  // 1/sqrt(64)

typedef __bf16 v16bf __attribute__((ext_vector_type(16)));
typedef float  v8f   __attribute__((ext_vector_type(8)));
typedef unsigned int v4u __attribute__((ext_vector_type(4)));
typedef int          v4i __attribute__((ext_vector_type(4)));
typedef int          v8i __attribute__((ext_vector_type(8)));

struct alignas(16) U4 { unsigned int x, y, z, w; };
struct alignas(16) F4 { float x, y, z, w; };

union Frag {
  v16bf v;
  unsigned short h[16];
  U4 q[2];
};

__device__ __forceinline__ unsigned short f2bf(float f) {
  union { __bf16 b; unsigned short u; } cv;
  cv.b = (__bf16)f;  // native v_cvt to bf16 (RNE)
  return cv.u;
}

__device__ __forceinline__ v8f wmma_bf16(const Frag& a, const Frag& b, v8f c) {
  // (neg_a, A, neg_b, B, c_mod, C, reuse_a, reuse_b)
  return __builtin_amdgcn_wmma_f32_16x16x32_bf16(false, a.v, false, b.v,
                                                 (short)0, c, false, false);
}

// --- TDM: stage a [rows=64][k=32] bf16 tile (row stride = ldK elements) into
// LDS at byte offset lds_addr.  Descriptor packed per CDNA5 ISA 8.3/8.4.
// Uniform (SGPR) arguments only; issued by one wave, tracked by TENSORcnt.
__device__ __forceinline__ void tdm_load_tile_64x32(
    const unsigned short* gsrc, unsigned lds_addr, unsigned ldK) {
  unsigned long long ga = (unsigned long long)(uintptr_t)gsrc;
  v4u g0;
  g0[0] = 1u;                                   // count=1, user mode
  g0[1] = lds_addr;                             // LDS byte address
  g0[2] = (unsigned)ga;                         // global_addr[31:0]
  g0[3] = (unsigned)(ga >> 32) | (2u << 30);    // global_addr[56:32] | type=2
  v8i g1;
  g1[0] = (int)(1u << 16);                      // wg_mask=0, data_size=1 (2B)
  g1[1] = (int)((ldK & 0xffffu) << 16);         // tensor_dim0[15:0]
  g1[2] = (int)(((ldK >> 16) & 0xffffu) | ((ldK & 0xffffu) << 16));
                                                // tensor_dim0[31:16] | tensor_dim1[15:0]
  g1[3] = (int)(((ldK >> 16) & 0xffffu) | (32u << 16));
                                                // tensor_dim1[31:16] | tile_dim0=32
  g1[4] = 64;                                   // tile_dim1=64, tile_dim2=0
  g1[5] = (int)ldK;                             // tensor_dim0_stride[31:0]
  g1[6] = 0;                                    // stride hi, tensor_dim1_stride lo
  g1[7] = 0;
  v4i z4 = {0, 0, 0, 0};
#if defined(__clang_major__) && (__clang_major__ >= 23)
  v8i z8 = {0, 0, 0, 0, 0, 0, 0, 0};
  __builtin_amdgcn_tensor_load_to_lds(g0, g1, z4, z4, z8, 0);
#else
  __builtin_amdgcn_tensor_load_to_lds(g0, g1, z4, z4, 0);
#endif
}

// ---------------------------------------------------------------------------
// Kernel 0: bulk fp32 -> bf16 conversion (8 elements/thread, b128 traffic).
// ---------------------------------------------------------------------------
__global__ __launch_bounds__(256) void cvt_bf16_kernel(
    const float* __restrict__ in, unsigned short* __restrict__ out, int n) {
  const int i = (blockIdx.x * 256 + threadIdx.x) * 8;
  if (i >= n) return;
  F4 a = ((const F4*)(in + i))[0];
  F4 b = ((const F4*)(in + i))[1];
  union { U4 q; unsigned short h[8]; } o;
  o.h[0] = f2bf(a.x); o.h[1] = f2bf(a.y); o.h[2] = f2bf(a.z); o.h[3] = f2bf(a.w);
  o.h[4] = f2bf(b.x); o.h[5] = f2bf(b.y); o.h[6] = f2bf(b.z); o.h[7] = f2bf(b.w);
  *(U4*)(out + i) = o.q;
}

// ---------------------------------------------------------------------------
// Kernel 1: QKV projections.  out = x @ W^T + b, head-major bf16 (V^T).
// grid = (M/128, D/64, 3); block = 256 (8 waves, each wave a 16x64 tile).
// Weight tile (64 cols x 32 K, 4 KB) TDM-staged into double-buffered LDS.
// ---------------------------------------------------------------------------
__global__ __launch_bounds__(256) void qkv_kernel(
    const unsigned short* __restrict__ xb,
    const unsigned short* __restrict__ wqb, const float* __restrict__ bq,
    const unsigned short* __restrict__ wkb, const float* __restrict__ bk,
    const unsigned short* __restrict__ wvb, const float* __restrict__ bv,
    unsigned short* __restrict__ qb, unsigned short* __restrict__ kb,
    unsigned short* __restrict__ vtb) {
  __shared__ unsigned short wtile[2][64 * 32];  // 2 x 4 KB
  const int lane = threadIdx.x & 31;
  const int wave = threadIdx.x >> 5;
  const int ln   = lane & 15;
  const int kb0    = (lane < 16) ? 0 : 16;
  const int kshift = (lane < 16) ? 0 : 16;
  const int half8  = (lane < 16) ? 0 : 8;
  const int z = blockIdx.z;

  const unsigned short* w = (z == 0) ? wqb : ((z == 1) ? wkb : wvb);
  const float* bias       = (z == 0) ? bq  : ((z == 1) ? bk  : bv);
  unsigned short* outp    = (z == 0) ? qb  : ((z == 1) ? kb  : vtb);

  const int m0 = blockIdx.x * 128 + wave * 16;  // rows in [0, B*S)
  const int n0 = blockIdx.y * 64;               // cols in [0, D)

  const unsigned ldsbase = (unsigned)(uintptr_t)&wtile[0][0];
  const unsigned short* wbase = w + (size_t)n0 * Dc;  // tile origin (64 rows)

  v8f acc[4];
#pragma unroll
  for (int t = 0; t < 4; ++t) acc[t] = 0.f;

  if (wave == 0) tdm_load_tile_64x32(wbase, ldsbase, Dc);

  const unsigned short* ar = xb + (size_t)(m0 + ln) * Dc;
  int p = 0;
  for (int k0 = 0; k0 < Dc; k0 += 32) {
    if (wave == 0) {
      if (k0 + 32 < Dc) {
        tdm_load_tile_64x32(wbase + k0 + 32,
                            ldsbase + (unsigned)((p ^ 1) * 64 * 32 * 2), Dc);
        __builtin_amdgcn_s_wait_tensorcnt(1);  // current buffer ready
      } else {
        __builtin_amdgcn_s_wait_tensorcnt(0);
      }
    }
    __syncthreads();

    __builtin_prefetch(ar + k0 + 256, 0, 1);  // global_prefetch_b8
    Frag a;
    a.q[0] = *(const U4*)(ar + k0 + kb0);
    a.q[1] = *(const U4*)(ar + k0 + kb0 + 8);
    const unsigned short* wt = &wtile[p][0];
#pragma unroll
    for (int t = 0; t < 4; ++t) {
      const unsigned short* br = wt + (t * 16 + ln) * 32 + kshift;  // ds_load
      Frag bfr;
      bfr.q[0] = ((const U4*)br)[0];
      bfr.q[1] = ((const U4*)br)[1];
      acc[t] = wmma_bf16(a, bfr, acc[t]);
    }
    __syncthreads();  // all waves done with buffer p before TDM rewrites it
    p ^= 1;
  }

#pragma unroll
  for (int t = 0; t < 4; ++t) {
    const int col = n0 + t * 16 + ln;
    const int hh = col >> 6;   // head
    const int hd = col & 63;
    const float bcol = bias[col];
#pragma unroll
    for (int r = 0; r < 8; ++r) {
      const int rw = m0 + r + half8;      // C layout: vgpr r -> row r / r+8
      const int bb = rw >> 11;            // / S
      const int ss = rw & (Sc - 1);
      float val = acc[t][r] + bcol;
      if (z == 0) val *= SCALEF;          // fold softmax scale into Q
      size_t idx;
      if (z == 2)  // V stored transposed: [bh][hd][s] for contiguous P@V loads
        idx = (((size_t)(bb * Hc + hh)) * HDc + hd) * Sc + ss;
      else
        idx = (((size_t)(bb * Hc + hh)) * Sc + ss) * HDc + hd;
      outp[idx] = f2bf(val);
    }
  }
}

// ---------------------------------------------------------------------------
// Kernel 2: fused flash attention.  Per wave: one 16-row q tile; kv loop in
// steps of 32 with online softmax; P re-shaped C->A layout through LDS.
// grid = (S/128, B*H); block = 256.
// ---------------------------------------------------------------------------
__global__ __launch_bounds__(256) void attn_kernel(
    const unsigned short* __restrict__ qb, const unsigned short* __restrict__ kbm,
    const unsigned short* __restrict__ vtb, const int* __restrict__ mask,
    unsigned short* __restrict__ ctxb) {
  __shared__ unsigned short pbuf[8 * 16 * 32];  // 1 KB per wave
  const int lane = threadIdx.x & 31;
  const int wave = threadIdx.x >> 5;
  const int ln   = lane & 15;
  const int kb0    = (lane < 16) ? 0 : 16;
  const int kshift = (lane < 16) ? 0 : 16;
  const int half8  = (lane < 16) ? 0 : 8;

  const int bh = blockIdx.y;
  const int b  = bh >> 4;
  const int h  = bh & 15;
  const int m0 = blockIdx.x * 128 + wave * 16;  // q row within S

  // Q A-fragments (16 x 64 = two 16x32 fragments), hoisted out of kv loop.
  const unsigned short* qrow = qb + ((size_t)bh * Sc + (m0 + ln)) * HDc;
  Frag qf[2];
#pragma unroll
  for (int c = 0; c < 2; ++c) {
    qf[c].q[0] = *(const U4*)(qrow + c * 32 + kb0);
    qf[c].q[1] = *(const U4*)(qrow + c * 32 + kb0 + 8);
  }

  v8f o[4];
#pragma unroll
  for (int t = 0; t < 4; ++t) o[t] = 0.f;
  float rm[8], rl[8];
#pragma unroll
  for (int r = 0; r < 8; ++r) { rm[r] = -1e30f; rl[r] = 0.f; }

  unsigned short* pw = pbuf + wave * (16 * 32);
  const int mbase = b * Sc;

  for (int kv0 = 0; kv0 < Sc; kv0 += 32) {
    // ---- scores: two 16x16 C tiles (kv columns kv0..+15, kv0+16..+31) ----
    v8f s0v = 0.f, s1v = 0.f;
#pragma unroll
    for (int c = 0; c < 2; ++c) {  // contraction over d in chunks of 32
      const unsigned short* kr0 =
          kbm + ((size_t)bh * Sc + (kv0 + ln)) * HDc + c * 32 + kshift;
      Frag bf0; bf0.q[0] = ((const U4*)kr0)[0]; bf0.q[1] = ((const U4*)kr0)[1];
      s0v = wmma_bf16(qf[c], bf0, s0v);
      const unsigned short* kr1 =
          kbm + ((size_t)bh * Sc + (kv0 + 16 + ln)) * HDc + c * 32 + kshift;
      Frag bf1; bf1.q[0] = ((const U4*)kr1)[0]; bf1.q[1] = ((const U4*)kr1)[1];
      s1v = wmma_bf16(qf[c], bf1, s1v);
    }

    // ---- key-padding mask + online softmax (per C-layout row r / r+8) ----
    const int mv0 = mask[mbase + kv0 + ln];
    const int mv1 = mask[mbase + kv0 + 16 + ln];
    float p0[8], p1[8], fac[8];
#pragma unroll
    for (int r = 0; r < 8; ++r) {
      float a0 = mv0 ? s0v[r] : -1e9f;
      float a1 = mv1 ? s1v[r] : -1e9f;
      float t = fmaxf(a0, a1);  // row max over 32 columns (one 16-lane half)
      t = fmaxf(t, __shfl_xor(t, 1, 32));
      t = fmaxf(t, __shfl_xor(t, 2, 32));
      t = fmaxf(t, __shfl_xor(t, 4, 32));
      t = fmaxf(t, __shfl_xor(t, 8, 32));
      const float mn = fmaxf(rm[r], t);
      fac[r] = __expf(rm[r] - mn);
      rm[r] = mn;
      const float e0 = __expf(a0 - mn);
      const float e1 = __expf(a1 - mn);
      float rs = e0 + e1;
      rs += __shfl_xor(rs, 1, 32);
      rs += __shfl_xor(rs, 2, 32);
      rs += __shfl_xor(rs, 4, 32);
      rs += __shfl_xor(rs, 8, 32);
      rl[r] = rl[r] * fac[r] + rs;
      p0[r] = e0; p1[r] = e1;
    }
#pragma unroll
    for (int t = 0; t < 4; ++t)
#pragma unroll
      for (int r = 0; r < 8; ++r) o[t][r] *= fac[r];

    // ---- P: C layout -> A layout via per-wave LDS (in-order DS unit) ----
#pragma unroll
    for (int r = 0; r < 8; ++r) {
      const int rloc = r + half8;
      pw[rloc * 32 + ln]      = f2bf(p0[r]);
      pw[rloc * 32 + 16 + ln] = f2bf(p1[r]);
    }
    asm volatile("" ::: "memory");
    Frag pf;
    pf.q[0] = *(const U4*)(pw + ln * 32 + kb0);
    pf.q[1] = *(const U4*)(pw + ln * 32 + kb0 + 8);
    asm volatile("" ::: "memory");

    // ---- ctx += P(16x32) @ V(32x64): V^T rows give contiguous B loads ----
#pragma unroll
    for (int t = 0; t < 4; ++t) {
      const unsigned short* vr =
          vtb + ((size_t)bh * HDc + (t * 16 + ln)) * Sc + kv0 + kshift;
      Frag bf; bf.q[0] = ((const U4*)vr)[0]; bf.q[1] = ((const U4*)vr)[1];
      o[t] = wmma_bf16(pf, bf, o[t]);
    }
  }

  // ---- normalize and store ctx back in [B,S,D] bf16 for the out-proj ----
#pragma unroll
  for (int t = 0; t < 4; ++t) {
#pragma unroll
    for (int r = 0; r < 8; ++r) {
      const int rw = m0 + r + half8;
      const float val = o[t][r] / fmaxf(rl[r], 1e-20f);
      ctxb[((size_t)b * Sc + rw) * Dc + h * 64 + t * 16 + ln] = f2bf(val);
    }
  }
}

// ---------------------------------------------------------------------------
// Kernel 3: output projection.  out = ctx @ wo^T + bo, fp32 result.
// grid = (M/128, D/64); block = 256.  Same TDM-staged weight path.
// ---------------------------------------------------------------------------
__global__ __launch_bounds__(256) void oproj_kernel(
    const unsigned short* __restrict__ ctxb, const unsigned short* __restrict__ wob,
    const float* __restrict__ bo, float* __restrict__ out) {
  __shared__ unsigned short wtile[2][64 * 32];
  const int lane = threadIdx.x & 31;
  const int wave = threadIdx.x >> 5;
  const int ln   = lane & 15;
  const int kb0    = (lane < 16) ? 0 : 16;
  const int kshift = (lane < 16) ? 0 : 16;
  const int half8  = (lane < 16) ? 0 : 8;

  const int m0 = blockIdx.x * 128 + wave * 16;
  const int n0 = blockIdx.y * 64;

  const unsigned ldsbase = (unsigned)(uintptr_t)&wtile[0][0];
  const unsigned short* wbase = wob + (size_t)n0 * Dc;

  v8f acc[4];
#pragma unroll
  for (int t = 0; t < 4; ++t) acc[t] = 0.f;

  if (wave == 0) tdm_load_tile_64x32(wbase, ldsbase, Dc);

  const unsigned short* ar = ctxb + (size_t)(m0 + ln) * Dc;
  int p = 0;
  for (int k0 = 0; k0 < Dc; k0 += 32) {
    if (wave == 0) {
      if (k0 + 32 < Dc) {
        tdm_load_tile_64x32(wbase + k0 + 32,
                            ldsbase + (unsigned)((p ^ 1) * 64 * 32 * 2), Dc);
        __builtin_amdgcn_s_wait_tensorcnt(1);
      } else {
        __builtin_amdgcn_s_wait_tensorcnt(0);
      }
    }
    __syncthreads();

    __builtin_prefetch(ar + k0 + 256, 0, 1);
    Frag a;
    a.q[0] = *(const U4*)(ar + k0 + kb0);
    a.q[1] = *(const U4*)(ar + k0 + kb0 + 8);
    const unsigned short* wt = &wtile[p][0];
#pragma unroll
    for (int t = 0; t < 4; ++t) {
      const unsigned short* br = wt + (t * 16 + ln) * 32 + kshift;
      Frag bfr;
      bfr.q[0] = ((const U4*)br)[0];
      bfr.q[1] = ((const U4*)br)[1];
      acc[t] = wmma_bf16(a, bfr, acc[t]);
    }
    __syncthreads();
    p ^= 1;
  }

#pragma unroll
  for (int t = 0; t < 4; ++t) {
    const int col = n0 + t * 16 + ln;
    const float bcol = bo[col];
#pragma unroll
    for (int r = 0; r < 8; ++r) {
      const int rw = m0 + r + half8;
      out[(size_t)rw * Dc + col] = acc[t][r] + bcol;
    }
  }
}

// ---------------------------------------------------------------------------
extern "C" void kernel_launch(void* const* d_in, const int* in_sizes, int n_in,
                              void* d_out, int out_size, void* d_ws,
                              size_t ws_size, hipStream_t stream) {
  const float* x  = (const float*)d_in[0];
  const int* mask = (const int*)d_in[1];
  const float* wq = (const float*)d_in[2];
  const float* bq = (const float*)d_in[3];
  const float* wk = (const float*)d_in[4];
  const float* bk = (const float*)d_in[5];
  const float* wv = (const float*)d_in[6];
  const float* bv = (const float*)d_in[7];
  const float* wo = (const float*)d_in[8];
  const float* bo = (const float*)d_in[9];
  float* out = (float*)d_out;

  const size_t NX = (size_t)Bc * Sc * Dc;  // 8,388,608
  const size_t NW = (size_t)Dc * Dc;       // 1,048,576
  unsigned short* xb   = (unsigned short*)d_ws;
  unsigned short* wqb  = xb + NX;
  unsigned short* wkb  = wqb + NW;
  unsigned short* wvb  = wkb + NW;
  unsigned short* wob  = wvb + NW;
  unsigned short* qb   = wob + NW;
  unsigned short* kb   = qb + NX;
  unsigned short* vtb  = kb + NX;
  unsigned short* ctxb = vtb + NX;  // total 92.3 MB

  dim3 blk(256);
  cvt_bf16_kernel<<<dim3((unsigned)(NX / 8 / 256)), blk, 0, stream>>>(x, xb, (int)NX);
  cvt_bf16_kernel<<<dim3((unsigned)(NW / 8 / 256)), blk, 0, stream>>>(wq, wqb, (int)NW);
  cvt_bf16_kernel<<<dim3((unsigned)(NW / 8 / 256)), blk, 0, stream>>>(wk, wkb, (int)NW);
  cvt_bf16_kernel<<<dim3((unsigned)(NW / 8 / 256)), blk, 0, stream>>>(wv, wvb, (int)NW);
  cvt_bf16_kernel<<<dim3((unsigned)(NW / 8 / 256)), blk, 0, stream>>>(wo, wob, (int)NW);

  qkv_kernel<<<dim3((Bc * Sc) / 128, Dc / 64, 3), blk, 0, stream>>>(
      xb, wqb, bq, wkb, bk, wvb, bv, qb, kb, vtb);
  attn_kernel<<<dim3(Sc / 128, Bc * Hc), blk, 0, stream>>>(
      qb, kb, vtb, mask, ctxb);
  oproj_kernel<<<dim3((Bc * Sc) / 128, Dc / 64), blk, 0, stream>>>(
      ctxb, wo ? wob : wob, bo, out);
}
